// GCL2_Loss_44796508897300
// MI455X (gfx1250) — matrix operand order repaired and checked by reading, not compiled, not run on hardware
//
#include <hip/hip_runtime.h>
#include <hip/hip_bf16.h>

typedef __attribute__((ext_vector_type(16))) _Float16 v16h;
typedef __attribute__((ext_vector_type(8)))  _Float16 v8h;
typedef __attribute__((ext_vector_type(8)))  float    v8f;

#define NROWS 8192
#define DDIM  128
#define EPSL  1e-8f

// ---------------------------------------------------------------------------
// Kernel 1: L2-normalize rows of f1/f2 (f32 -> f16). One wave per row.
// ---------------------------------------------------------------------------
__global__ __launch_bounds__(256) void nrm_kernel(
    const float* __restrict__ f1, const float* __restrict__ f2,
    _Float16* __restrict__ g1, _Float16* __restrict__ g2) {
  const int wave = threadIdx.x >> 5;
  const int lane = threadIdx.x & 31;
  const int row  = blockIdx.x * 8 + wave;
  if (row >= NROWS) return;

  const float* srcs[2] = {f1, f2};
  _Float16*    dsts[2] = {g1, g2};
#pragma unroll
  for (int s = 0; s < 2; ++s) {
    const float* r = srcs[s] + (size_t)row * DDIM;
    float x0 = r[lane], x1 = r[lane + 32], x2 = r[lane + 64], x3 = r[lane + 96];
    float ss = x0 * x0 + x1 * x1 + x2 * x2 + x3 * x3;
#pragma unroll
    for (int off = 16; off >= 1; off >>= 1) ss += __shfl_xor(ss, off, 32);
    float inv = 1.0f / fmaxf(sqrtf(ss), 1e-12f);
    _Float16* w = dsts[s] + (size_t)row * DDIM;
    w[lane]      = (_Float16)(x0 * inv);
    w[lane + 32] = (_Float16)(x1 * inv);
    w[lane + 64] = (_Float16)(x2 * inv);
    w[lane + 96] = (_Float16)(x3 * inv);
  }
}

// ---------------------------------------------------------------------------
// Kernel 2: main similarity-reduction kernel.
// One workgroup (4 waves) per 16-row block; waves split the 512 column tiles.
// Per 16x16 tile: 12x v_wmma_f32_16x16x32_f16 (sim12, sim11, sim22 over K=128),
// exp, mask-weighted accumulation into 7 per-row partial sums.
// ---------------------------------------------------------------------------
__global__ __launch_bounds__(128) void simloss_main(
    const _Float16* __restrict__ g1, const _Float16* __restrict__ g2,
    const float* __restrict__ mask, float* __restrict__ rowacc) {
  const int wave  = threadIdx.x >> 5;
  const int lane  = threadIdx.x & 31;
  const int hi    = lane >> 4;   // half-wave select
  const int ln    = lane & 15;
  const int ibase = blockIdx.x * 16;

  // --- Load loop-invariant A tiles (rows ibase..ibase+15, K=0..127) --------
  // 16-bit A 16x32 layout: lane m=0..15 (x2 halves), VGPR v holds
  // K = hi*16 + v*2 (+0,+1) within each 32-wide K chunk (matches B).
  v16h A1[4], A2[4];
  {
    const _Float16* r1 = g1 + (size_t)(ibase + ln) * DDIM;
    const _Float16* r2 = g2 + (size_t)(ibase + ln) * DDIM;
#pragma unroll
    for (int kc = 0; kc < 4; ++kc) {
      const int k0 = kc * 32 + hi * 16;      // halves 0..7
      const int k1 = kc * 32 + hi * 16 + 8;  // halves 8..15
      v8h lo1 = *(const v8h*)(r1 + k0), up1 = *(const v8h*)(r1 + k1);
      v8h lo2 = *(const v8h*)(r2 + k0), up2 = *(const v8h*)(r2 + k1);
#pragma unroll
      for (int t = 0; t < 8; ++t) {
        A1[kc][t] = lo1[t]; A1[kc][t + 8] = up1[t];
        A2[kc][t] = lo2[t]; A2[kc][t + 8] = up2[t];
      }
    }
  }

  // acc[0]=s12*m  acc[1]=s12  acc[2]=s11*m  acc[3]=s11
  // acc[4]=s22*m  acc[5]=s22  acc[6]=mask sum
  float acc[7][8] = {};

  for (int jt = wave; jt < NROWS / 16; jt += 4) {
    const int jbase = jt * 16;

    // --- B tiles: contiguous k order (k = hi*16 + s) -> 16 consecutive halves/lane, same as A
    const _Float16* c1 = g1 + (size_t)(jbase + ln) * DDIM + hi * 16;
    const _Float16* c2 = g2 + (size_t)(jbase + ln) * DDIM + hi * 16;
    v16h B1[4], B2[4];
#pragma unroll
    for (int kc = 0; kc < 4; ++kc) {
      B1[kc] = *(const v16h*)(c1 + kc * 32);
      B2[kc] = *(const v16h*)(c2 + kc * 32);
    }

    // Prefetch this wave's next mask tile (stride 4 tiles = 64 columns).
    if (jbase + 64 < NROWS)
      __builtin_prefetch(mask + (size_t)(ibase + ln) * NROWS + jbase + 64 + hi * 8, 0, 0);

    // --- three 16x16 products over K=128 --------------------------------
    v8f c12 = {}, c11 = {}, c22 = {};
#pragma unroll
    for (int kc = 0; kc < 4; ++kc) {
      c12 = __builtin_amdgcn_wmma_f32_16x16x32_f16(false, A1[kc], false, B2[kc],
                                                   (short)0, c12, false, false);
      c11 = __builtin_amdgcn_wmma_f32_16x16x32_f16(false, A1[kc], false, B1[kc],
                                                   (short)0, c11, false, false);
      c22 = __builtin_amdgcn_wmma_f32_16x16x32_f16(false, A2[kc], false, B2[kc],
                                                   (short)0, c22, false, false);
    }

    // --- exp + masked accumulation (C layout: M = v + 8*hi, N = ln) ------
    const float* mrow = mask + (size_t)(ibase + hi * 8) * NROWS + jbase + ln;
#pragma unroll
    for (int v = 0; v < 8; ++v) {
      float m = __builtin_nontemporal_load(mrow + (size_t)v * NROWS);
      float e12 = __expf(c12[v]);
      float e11 = __expf(c11[v]);
      float e22 = __expf(c22[v]);
      acc[0][v] = fmaf(e12, m, acc[0][v]);  acc[1][v] += e12;
      acc[2][v] = fmaf(e11, m, acc[2][v]);  acc[3][v] += e11;
      acc[4][v] = fmaf(e22, m, acc[4][v]);  acc[5][v] += e22;
      acc[6][v] += m;
    }
  }

  // --- reduce over N within each 16-lane half ----------------------------
#pragma unroll
  for (int off = 8; off >= 1; off >>= 1) {
#pragma unroll
    for (int a = 0; a < 7; ++a)
#pragma unroll
      for (int v = 0; v < 8; ++v)
        acc[a][v] += __shfl_xor(acc[a][v], off, 32);
  }

  // --- combine the 4 waves via LDS float atomics -------------------------
  __shared__ float red[7][16];
  if (threadIdx.x < 112) ((float*)red)[threadIdx.x] = 0.0f;
  __syncthreads();
  if (ln == 0) {
#pragma unroll
    for (int a = 0; a < 7; ++a)
#pragma unroll
      for (int v = 0; v < 8; ++v)
        atomicAdd(&red[a][hi * 8 + v], acc[a][v]);
  }
  __syncthreads();
  if (threadIdx.x < 16) {
#pragma unroll
    for (int a = 0; a < 7; ++a)
      rowacc[(size_t)a * NROWS + ibase + threadIdx.x] = red[a][threadIdx.x];
  }
}

// ---------------------------------------------------------------------------
// Kernel 3: finalize — per-row log terms (with exact diagonal patch), reduce.
// ---------------------------------------------------------------------------
__global__ __launch_bounds__(256) void fin_kernel(
    const float* __restrict__ rowacc, const float* __restrict__ mask,
    float* __restrict__ out) {
  const int i = blockIdx.x * blockDim.x + threadIdx.x;
  float contrib = 0.0f;
  if (i < NROWS) {
    float s12p = rowacc[0 * NROWS + i], s12t = rowacc[1 * NROWS + i];
    float s11p = rowacc[2 * NROWS + i], s11t = rowacc[3 * NROWS + i];
    float s22p = rowacc[4 * NROWS + i], s22t = rowacc[5 * NROWS + i];
    float msum = rowacc[6 * NROWS + i];
    float md   = mask[(size_t)i * NROWS + i];
    const float E = 2.7182818284590452f;  // exp(1/TEMPERATURE), sim[i,i]=1
    float pos1 = s12p + (s11p - md * E), tot1 = s12t + (s11t - E);
    float pos2 = s12p + (s22p - md * E), tot2 = s12t + (s22t - E);
    float denom = 2.0f * msum - md;       // sum(mask) + sum(offdiag mask)
    contrib = (logf((pos1 + EPSL) / (tot1 + EPSL)) +
               logf((pos2 + EPSL) / (tot2 + EPSL))) / denom;
  }
  __shared__ float sh[256];
  sh[threadIdx.x] = contrib;
  __syncthreads();
#pragma unroll
  for (int s = 128; s > 0; s >>= 1) {
    if (threadIdx.x < s) sh[threadIdx.x] += sh[threadIdx.x + s];
    __syncthreads();
  }
  if (threadIdx.x == 0) atomicAdd(out, -sh[0] / (2.0f * NROWS));
}

__global__ void zero_out_kernel(float* out) {
  if (threadIdx.x == 0) out[0] = 0.0f;
}

// ---------------------------------------------------------------------------
extern "C" void kernel_launch(void* const* d_in, const int* in_sizes, int n_in,
                              void* d_out, int out_size, void* d_ws, size_t ws_size,
                              hipStream_t stream) {
  const float* f1   = (const float*)d_in[0];
  const float* f2   = (const float*)d_in[1];
  const float* mask = (const float*)d_in[2];

  _Float16* g1     = (_Float16*)d_ws;                  // 8192*128 f16
  _Float16* g2     = g1 + (size_t)NROWS * DDIM;        // 8192*128 f16
  float*    rowacc = (float*)(g2 + (size_t)NROWS * DDIM);  // 7*8192 f32
  float*    out    = (float*)d_out;

  zero_out_kernel<<<1, 32, 0, stream>>>(out);
  nrm_kernel<<<NROWS / 8, 256, 0, stream>>>(f1, f2, g1, g2);
  simloss_main<<<NROWS / 16, 128, 0, stream>>>(g1, g2, mask, rowacc);
  fin_kernel<<<NROWS / 256, 256, 0, stream>>>(rowacc, mask, out);
}
